// DNF_Netowrk3_71159018160597
// MI455X (gfx1250) — matrix operand order repaired and checked by reading, not compiled, not run on hardware
//
#include <hip/hip_runtime.h>
#include <hip/hip_bf16.h>

typedef __attribute__((ext_vector_type(16))) _Float16 v16h;
typedef __attribute__((ext_vector_type(8)))  float    v8f;

#define B_   256
#define I_   512
#define H_   1024
#define O_   128
#define KC   8
#define NPOW 16                 // Mercator terms: log(1-z) = -sum z^n/n

// fragment-plane geometry
#define A_PLANE 8192            // halves per (n,bt) A plane: 16 blk * 32 lane * 16 h
#define B_PLANE 32768           // halves per (n,hs) B plane: 16 blk * 4 wv * 32 lane * 16 h
#define U_STRIDE (16 * A_PLANE) // halves per n over all 16 b-tiles  = 131072
#define W_STRIDE (16 * B_PLANE) // halves per n over all 16 h-slabs  = 524288
#define LDSBUF (A_PLANE + B_PLANE)   // 40960 halves = 80 KB per buffer

#define USE_ASYNC 1             // gfx1250 async-to-LDS path (inline asm)

// ---------------------------------------------------------------------------
// Raise kernels: write fragment-ordered f16 planes of u^n / clip(Wa)^n to
// global. Power chain kept in f32. One thread per fragment slot; the (lane,h)
// -> (M,K) un-swizzle implements the ISA 7.12.2 16-bit A / B layouts so the
// GEMM can load v16h fragments as two contiguous ds_load_b128.
// ---------------------------------------------------------------------------
__global__ void raise_u(const float* __restrict__ x, _Float16* __restrict__ U)
{
    const int idx   = blockIdx.x * 256 + threadIdx.x;        // 131072 slots
    const int h     = idx & 15;
    const int lane  = (idx >> 4) & 31;
    const int block = (idx >> 9) & 15;
    const int bt    = idx >> 13;
    const int M     = lane & 15;
    const int K     = (lane < 16) ? ((h < 8) ? h : h + 8)
                                  : ((h < 8) ? h + 8 : h + 16);
    const float base = 1.0f - x[(bt * 16 + M) * I_ + block * 32 + K];   // u = 1-x
    const int pos = bt * A_PLANE + (block * 32 + lane) * 16 + h;
    float p = base;
    #pragma unroll
    for (int n = 0; n < NPOW; ++n) {
        U[n * U_STRIDE + pos] = (_Float16)p;
        p *= base;
    }
}

__global__ void raise_w(const float* __restrict__ Wa, _Float16* __restrict__ W)
{
    const int idx   = blockIdx.x * 256 + threadIdx.x;        // 524288 slots
    const int h     = idx & 15;
    const int lane  = (idx >> 4) & 31;
    const int wv    = (idx >> 9) & 3;
    const int block = (idx >> 11) & 15;
    const int hs    = idx >> 15;
    const int N     = lane & 15;
    const int K     = ((lane < 16) ? 0 : 8) + ((h < 8) ? h : h + 8);
    float w = Wa[(block * 32 + K) * H_ + hs * 64 + wv * 16 + N];
    w = fminf(fmaxf(w, 0.0f), 1.0f);                          // clip to [0,1]
    const int pos = hs * B_PLANE + ((block * 4 + wv) * 32 + lane) * 16 + h;
    float p = w;
    #pragma unroll
    for (int n = 0; n < NPOW; ++n) {
        W[n * W_STRIDE + pos] = (_Float16)p;
        p *= w;
    }
}

// ---------------------------------------------------------------------------
// Async staging: copy one (A,B) fragment plane pair (80 KB) global -> LDS.
// 40 global_load_async_to_lds_b128 per thread (8 for A, 32 for B).
// ---------------------------------------------------------------------------
__device__ __forceinline__ void kick_plane(const _Float16* __restrict__ Up,
                                           const _Float16* __restrict__ Wp,
                                           _Float16* buf, int tid)
{
#if USE_ASYNC
    #pragma unroll
    for (int j = 0; j < 8; ++j) {
        const unsigned d = (unsigned)(size_t)(buf + (tid + j * 128) * 8);
        const _Float16* g = Up + (tid + j * 128) * 8;
        asm volatile("global_load_async_to_lds_b128 %0, %1, off"
                     :: "v"(d), "v"(g) : "memory");
    }
    #pragma unroll
    for (int j = 0; j < 32; ++j) {
        const unsigned d = (unsigned)(size_t)(buf + A_PLANE + (tid + j * 128) * 8);
        const _Float16* g = Wp + (tid + j * 128) * 8;
        asm volatile("global_load_async_to_lds_b128 %0, %1, off"
                     :: "v"(d), "v"(g) : "memory");
    }
#else
    #pragma unroll
    for (int j = 0; j < 8; ++j)
        ((float4*)buf)[tid + j * 128] = ((const float4*)Up)[tid + j * 128];
    #pragma unroll
    for (int j = 0; j < 32; ++j)
        ((float4*)(buf + A_PLANE))[tid + j * 128] = ((const float4*)Wp)[tid + j * 128];
#endif
}

#if USE_ASYNC
#define WAIT_ASYNC(N) asm volatile("s_wait_asynccnt %0" :: "i"(N))
#else
#define WAIT_ASYNC(N)
#endif

// ---------------------------------------------------------------------------
// GEMM: S[b,h] = -sum_n (1/n) * (U^n @ W^n)[b,h] = sum_i log(1 - u*W).
// WG = 4 waves, 16(b) x 64(h) tile; power planes double-buffered in LDS via
// async copies; inner loop is 2+2 ds_load_b128 + one v_wmma_f32_16x16x32_f16.
// ---------------------------------------------------------------------------
__global__ void __launch_bounds__(128)
dnf_gemm(const _Float16* __restrict__ U, const _Float16* __restrict__ W,
         float* __restrict__ S)
{
    extern __shared__ _Float16 ldsh[];           // 2 * LDSBUF halves (160 KB)

    const int tid  = threadIdx.x;
    const int lane = tid & 31;
    const int wave = tid >> 5;
    const int bt   = blockIdx.x;                 // 16 b-tiles
    const int hs   = blockIdx.y;                 // 16 h-slabs
    const int b0   = bt * 16;
    const int h0   = hs * 64 + wave * 16;
    const int mrow = lane & 15;
    const bool hi  = lane >= 16;

    const _Float16* Ub = U + bt * A_PLANE;       // + n*U_STRIDE per power
    const _Float16* Wb = W + hs * B_PLANE;       // + n*W_STRIDE per power

    v8f total = {0.f, 0.f, 0.f, 0.f, 0.f, 0.f, 0.f, 0.f};

    kick_plane(Ub, Wb, ldsh, tid);               // preload n=1 into buffer 0

    for (int n = 1; n <= NPOW; ++n) {
        const int cur = (n - 1) & 1;
        if (n < NPOW)                            // prefetch n+1 into other buffer
            kick_plane(Ub + n * U_STRIDE, Wb + n * W_STRIDE,
                       ldsh + (cur ^ 1) * LDSBUF, tid);
        if (n < NPOW) { WAIT_ASYNC(40); } else { WAIT_ASYNC(0); }  // plane n done
        __syncthreads();

        const _Float16* Ab = ldsh + cur * LDSBUF;
        const _Float16* Bb = Ab + A_PLANE;

        v8f acc = {0.f, 0.f, 0.f, 0.f, 0.f, 0.f, 0.f, 0.f};
        #pragma unroll
        for (int j = 0; j < 16; ++j) {           // K = i dimension, steps of 32
            const v16h a = *(const v16h*)(Ab + (j * 32 + lane) * 16);
            const v16h b = *(const v16h*)(Bb + ((j * 4 + wave) * 32 + lane) * 16);
            acc = __builtin_amdgcn_wmma_f32_16x16x32_f16(
                      false, a, false, b, (short)0, acc, false, false);
        }
        const float sa = 1.0f / (float)n;        // series coefficient
        #pragma unroll
        for (int r = 0; r < 8; ++r) total[r] += sa * acc[r];
        __syncthreads();                         // buffer reusable after this
    }

    // C/D layout: VGPR r -> M = r (lanes 0-15) / r+8 (lanes 16-31), N = lane%16
    #pragma unroll
    for (int r = 0; r < 8; ++r) {
        const int m = hi ? r + 8 : r;
        S[(b0 + m) * H_ + h0 + mrow] = -total[r];
    }
}

// ---------------------------------------------------------------------------
// Fuzzy OR over the K=8 conjunctions: out[b,o] = 1 - prod_k(1 - Wo * e^S)
// ---------------------------------------------------------------------------
__global__ void dnf_or(const float* __restrict__ S, const float* __restrict__ Wo,
                       float* __restrict__ out)
{
    const int idx = blockIdx.x * blockDim.x + threadIdx.x;
    if (idx >= B_ * O_) return;
    const int b = idx >> 7;
    const int o = idx & (O_ - 1);
    float p = 1.0f;
    #pragma unroll
    for (int k = 0; k < KC; ++k) {
        float w = Wo[o * KC + k];
        w = fminf(fmaxf(w, 0.0f), 1.0f);
        const float andv = __expf(S[b * H_ + o * KC + k]);   // S <= 0
        p *= (1.0f - w * andv);
    }
    out[idx] = 1.0f - p;
}

// ---------------------------------------------------------------------------
extern "C" void kernel_launch(void* const* d_in, const int* in_sizes, int n_in,
                              void* d_out, int out_size, void* d_ws, size_t ws_size,
                              hipStream_t stream)
{
    (void)in_sizes; (void)n_in; (void)out_size; (void)ws_size;

    const float* x  = (const float*)d_in[0];     // (256, 512, 1) f32
    const float* Wa = (const float*)d_in[1];     // (512, 1024)   f32
    const float* Wo = (const float*)d_in[2];     // (1024, 1)     f32
    float* out = (float*)d_out;                  // (256, 128)    f32

    // workspace: S 1MB | U planes 4MB | W planes 16MB  (L2-resident, 192MB L2)
    float*    S = (float*)d_ws;
    _Float16* U = (_Float16*)((char*)d_ws + (1u << 20));
    _Float16* W = (_Float16*)((char*)d_ws + 5u * (1u << 20));

    raise_u<<<(B_ * I_) / 256, 256, 0, stream>>>(x, U);
    raise_w<<<(I_ * H_) / 256, 256, 0, stream>>>(Wa, W);

    const size_t lds_bytes = 2u * LDSBUF * sizeof(_Float16);   // 160 KB
    (void)hipFuncSetAttribute((const void*)dnf_gemm,
                              hipFuncAttributeMaxDynamicSharedMemorySize,
                              (int)lds_bytes);
    dnf_gemm<<<dim3(16, 16), 128, lds_bytes, stream>>>(U, W, S);
    dnf_or<<<(B_ * O_ + 255) / 256, 256, 0, stream>>>(S, Wo, out);
}
